// LowrankLearnableHash_29987461661149
// MI455X (gfx1250) — compile-verified, matrix operand order checked by reading, not run on hardware
//
#include <hip/hip_runtime.h>
#include <hip/hip_bf16.h>

#define RAYS    8192
#define SAMPLES 128

typedef __attribute__((ext_vector_type(16))) _Float16 v16h;
typedef __attribute__((ext_vector_type(8)))  float    v8f;
typedef __attribute__((ext_vector_type(4)))  float    v4f;

// Load a 16x16x32-f16 WMMA operand (A or B) from an LDS tile stored row-major
// as [row][K] halves. CDNA5 16-bit operand layout: lanes 0-15 hold row=lane,
// K = kofs+{0..7, 16..23}; lanes 16-31 hold row=lane-16, K = kofs+{8..15, 24..31}.
// Row stride must keep 16B alignment (stride multiple of 8 halves).
__device__ __forceinline__ v16h lds_tile16(const _Float16* base, int stride_halves,
                                           int kofs, int lane) {
  const int row = lane & 15;
  const int kb  = kofs + ((lane >> 4) << 3);
  const _Float16* p = base + row * stride_halves + kb;
  v16h r;
  ((v4f*)&r)[0] = *(const v4f*)(p);
  ((v4f*)&r)[1] = *(const v4f*)(p + 16);
  return r;
}

__device__ __forceinline__ void bilerp3(const float* __restrict__ g,
                                        float ca, float cb, float out[3]) {
  // plane: [3][128][128], align_corners=True, border clamp
  float pa = fminf(fmaxf((ca + 1.f) * 63.5f, 0.f), 127.f);
  float pb = fminf(fmaxf((cb + 1.f) * 63.5f, 0.f), 127.f);
  int ia = (int)floorf(pa); ia = ia < 0 ? 0 : (ia > 126 ? 126 : ia);
  int ib = (int)floorf(pb); ib = ib < 0 ? 0 : (ib > 126 ? 126 : ib);
  float fa = pa - (float)ia, fb = pb - (float)ib;
  float w00 = (1.f - fa) * (1.f - fb), w01 = (1.f - fa) * fb;
  float w10 = fa * (1.f - fb),         w11 = fa * fb;
  int base = ia * 128 + ib;
#pragma unroll
  for (int c = 0; c < 3; ++c) {
    const float* p = g + c * 16384 + base;
    out[c] = p[0] * w00 + p[1] * w01 + p[128] * w10 + p[129] * w11;
  }
}

__global__ __launch_bounds__(256) void nerf_points_kernel(
    const float* __restrict__ rays_o, const float* __restrict__ rays_d,
    const float* __restrict__ p01, const float* __restrict__ p02,
    const float* __restrict__ p12, const float* __restrict__ fgrid,
    const float* __restrict__ w1, const float* __restrict__ b1,
    const float* __restrict__ w2, const float* __restrict__ b2,
    const float* __restrict__ wc1, const float* __restrict__ bc1,
    const float* __restrict__ wc2, const float* __restrict__ bc2,
    const float* __restrict__ aabb,
    float* __restrict__ density_out,   // [S][R]
    float* __restrict__ rgb_out)       // [3][S][R]
{
  __shared__ __align__(16) _Float16 Afeat[16][16][32];  // 16 sample-tiles, A operand
  __shared__ __align__(16) _Float16 Ac[16][16][32];     // color-MLP input tiles (K padded 18->32)
  __shared__ __align__(16) _Float16 Hh[8][16][64];      // per-wave hidden activations
  __shared__ __align__(16) _Float16 W1h[64][32];        // [N][K] transposed weights
  __shared__ __align__(16) _Float16 W2h[16][64];
  __shared__ __align__(16) _Float16 WC1h[64][32];
  __shared__ __align__(16) _Float16 WC2h[16][64];
  __shared__ float B1s[64], B2s[16], BC1s[64], BC2s[16];

  const int t = threadIdx.x;

  // ---- stage weights (transpose to [N][K] f16, zero-pad) ----
  for (int i = t; i < 64 * 32; i += 256) {
    int n = i >> 5, k = i & 31;
    W1h[n][k]  = (_Float16)w1[k * 64 + n];
    WC1h[n][k] = (k < 18) ? (_Float16)wc1[k * 64 + n] : (_Float16)0.f;
  }
  for (int i = t; i < 16 * 64; i += 256) {
    int n = i >> 6, k = i & 63;
    W2h[n][k]  = (_Float16)w2[k * 16 + n];
    WC2h[n][k] = (n < 3) ? (_Float16)wc2[k * 3 + n] : (_Float16)0.f;
  }
  if (t < 64) { B1s[t] = b1[t]; BC1s[t] = bc1[t]; }
  if (t < 16) { B2s[t] = b2[t]; BC2s[t] = (t < 3) ? bc2[t] : 0.f; }

  // ---- per-thread phase: one sample per thread ----
  const int n   = blockIdx.x * 256 + t;
  const int ray = n >> 7;           // SAMPLES = 128
  const int smp = n & 127;

  float dx = rays_d[ray * 3 + 0], dy = rays_d[ray * 3 + 1], dz = rays_d[ray * 3 + 2];
  float rn = rsqrtf(dx * dx + dy * dy + dz * dz);
  dx *= rn; dy *= rn; dz *= rn;
  float ox = rays_o[ray * 3 + 0], oy = rays_o[ray * 3 + 1], oz = rays_o[ray * 3 + 2];
  float a0x = aabb[0], a0y = aabb[1], a0z = aabb[2];
  float a1x = aabb[3], a1y = aabb[4], a1z = aabb[5];

  float ivx = 1.f / dx, ivy = 1.f / dy, ivz = 1.f / dz;
  float tx0 = (a0x - ox) * ivx, tx1 = (a1x - ox) * ivx;
  float ty0 = (a0y - oy) * ivy, ty1 = (a1y - oy) * ivy;
  float tz0 = (a0z - oz) * ivz, tz1 = (a1z - oz) * ivz;
  float nearv = fmaxf(fmaxf(fminf(tx0, tx1), fminf(ty0, ty1)), fminf(tz0, tz1));
  nearv = fmaxf(nearv, 0.f);
  float farv = fminf(fminf(fmaxf(tx0, tx1), fmaxf(ty0, ty1)), fmaxf(tz0, tz1));
  farv = fmaxf(farv, nearv);
  float delta = (farv - nearv) * (1.f / (float)SAMPLES);
  float tpar  = nearv + delta * ((float)smp + 0.5f);

  float px = ox + dx * tpar, py = oy + dy * tpar, pz = oz + dz * tpar;
  float nx = (px - a0x) * (2.f / (a1x - a0x)) - 1.f;
  float ny = (py - a0y) * (2.f / (a1y - a0y)) - 1.f;
  float nz = (pz - a0z) * (2.f / (a1z - a0z)) - 1.f;

  float i01[3], i02[3], i12[3];
  bilerp3(p01, nx, ny, i01);
  bilerp3(p02, nx, nz, i02);
  bilerp3(p12, ny, nz, i12);
  float cx = i01[0] * i02[0] * i12[0];
  float cy = i01[1] * i02[1] * i12[1];
  float cz = i01[2] * i02[2] * i12[2];

  // trilinear into [32][64][64][64] (L2-resident)
  float qx = fminf(fmaxf((cx + 1.f) * 31.5f, 0.f), 63.f);
  float qy = fminf(fmaxf((cy + 1.f) * 31.5f, 0.f), 63.f);
  float qz = fminf(fmaxf((cz + 1.f) * 31.5f, 0.f), 63.f);
  int ix = (int)floorf(qx); ix = ix < 0 ? 0 : (ix > 62 ? 62 : ix);
  int iy = (int)floorf(qy); iy = iy < 0 ? 0 : (iy > 62 ? 62 : iy);
  int iz = (int)floorf(qz); iz = iz < 0 ? 0 : (iz > 62 ? 62 : iz);
  float fx = qx - (float)ix, fy = qy - (float)iy, fz = qz - (float)iz;
  float wx[2] = {1.f - fx, fx}, wy[2] = {1.f - fy, fy}, wz[2] = {1.f - fz, fz};
  int base = (ix * 64 + iy) * 64 + iz;

  float feats[32];
#pragma unroll
  for (int c = 0; c < 32; ++c) feats[c] = 0.f;
#pragma unroll
  for (int c8 = 0; c8 < 8; ++c8) {
    int oxi = c8 >> 2, oyi = (c8 >> 1) & 1, ozi = c8 & 1;
    float w = wx[oxi] * wy[oyi] * wz[ozi];
    const float* p = fgrid + base + oxi * 4096 + oyi * 64 + ozi;
#pragma unroll
    for (int c = 0; c < 32; ++c) feats[c] += p[c * 262144] * w;
  }

  const int g = t >> 4, M = t & 15;
#pragma unroll
  for (int k = 0; k < 32; ++k) Afeat[g][M][k] = (_Float16)feats[k];
  Ac[g][M][0] = (_Float16)dx;
  Ac[g][M][1] = (_Float16)dy;
  Ac[g][M][2] = (_Float16)dz;
#pragma unroll
  for (int k = 18; k < 32; ++k) Ac[g][M][k] = (_Float16)0.f;

  __syncthreads();

  // ---- WMMA phase: each wave owns 2 sample-tiles ----
  const int wave = t >> 5;
  const int lane = t & 31;
  const int col  = lane & 15;
  const int mhi  = (lane >> 4) << 3;

  // Hoist ALL loop-invariant weight B operands into registers (96 VGPRs),
  // so the per-tile loop issues WMMAs without waiting on weight ds_loads.
  v16h bW1[4], bC1[4], bW2[2], bC2[2];
#pragma unroll
  for (int nb = 0; nb < 4; ++nb) {
    bW1[nb] = lds_tile16(&W1h[nb * 16][0], 32, 0, lane);
    bC1[nb] = lds_tile16(&WC1h[nb * 16][0], 32, 0, lane);
  }
#pragma unroll
  for (int j = 0; j < 2; ++j) {
    bW2[j] = lds_tile16(&W2h[0][0], 64, j * 32, lane);
    bC2[j] = lds_tile16(&WC2h[0][0], 64, j * 32, lane);
  }
  const float bb2  = B2s[col];
  const float bbc2 = (col < 3) ? BC2s[col] : 0.f;
  float bb1[4], bbc1[4];
#pragma unroll
  for (int nb = 0; nb < 4; ++nb) {
    bb1[nb]  = B1s[nb * 16 + col];
    bbc1[nb] = BC1s[nb * 16 + col];
  }

#pragma unroll
  for (int iter = 0; iter < 2; ++iter) {
    const int g2 = wave * 2 + iter;

    // sigma layer 1: [16x32] x [32x64]
    v16h a = lds_tile16(&Afeat[g2][0][0], 32, 0, lane);
    v8f h[4];
#pragma unroll
    for (int nb = 0; nb < 4; ++nb) {
      v8f zc = {};
      h[nb] = __builtin_amdgcn_wmma_f32_16x16x32_f16(false, a, false, bW1[nb],
                                                     (short)0, zc, false, false);
    }
#pragma unroll
    for (int nb = 0; nb < 4; ++nb) {
#pragma unroll
      for (int r = 0; r < 8; ++r) {
        float v = h[nb][r] + bb1[nb];
        Hh[wave][r + mhi][nb * 16 + col] = (_Float16)(v > 0.f ? v : 0.f);
      }
    }

    // sigma layer 2: [16x64] x [64x16], K split into two 32-blocks
    v8f sacc = {};
#pragma unroll
    for (int j = 0; j < 2; ++j) {
      v16h a2 = lds_tile16(&Hh[wave][0][0], 64, j * 32, lane);
      sacc = __builtin_amdgcn_wmma_f32_16x16x32_f16(false, a2, false, bW2[j],
                                                    (short)0, sacc, false, false);
    }
#pragma unroll
    for (int r = 0; r < 8; ++r) {
      float v = sacc[r] + bb2;
      int m = r + mhi;
      int nglob = blockIdx.x * 256 + g2 * 16 + m;
      int rr = nglob >> 7, ss = nglob & 127;
      if (col == 0) {
        density_out[ss * RAYS + rr] = __expf(fminf(fmaxf(v, -15.f), 15.f));
      } else {
        Ac[g2][m][2 + col] = (_Float16)v;  // cin[3 + (col-1)]
      }
    }

    // color layer 1: [16x32(pad 18)] x [32x64]
    v16h ac = lds_tile16(&Ac[g2][0][0], 32, 0, lane);
    v8f hc[4];
#pragma unroll
    for (int nb = 0; nb < 4; ++nb) {
      v8f zc = {};
      hc[nb] = __builtin_amdgcn_wmma_f32_16x16x32_f16(false, ac, false, bC1[nb],
                                                      (short)0, zc, false, false);
    }
#pragma unroll
    for (int nb = 0; nb < 4; ++nb) {
#pragma unroll
      for (int r = 0; r < 8; ++r) {
        float v = hc[nb][r] + bbc1[nb];
        Hh[wave][r + mhi][nb * 16 + col] = (_Float16)(v > 0.f ? v : 0.f);
      }
    }

    // color layer 2: [16x64] x [64x16(pad 3)]
    v8f oacc = {};
#pragma unroll
    for (int j = 0; j < 2; ++j) {
      v16h a2 = lds_tile16(&Hh[wave][0][0], 64, j * 32, lane);
      oacc = __builtin_amdgcn_wmma_f32_16x16x32_f16(false, a2, false, bC2[j],
                                                    (short)0, oacc, false, false);
    }
    if (col < 3) {
#pragma unroll
      for (int r = 0; r < 8; ++r) {
        float v = oacc[r] + bbc2;
        float sg = 1.f / (1.f + __expf(-v));
        int m = r + mhi;
        int nglob = blockIdx.x * 256 + g2 * 16 + m;
        int rr = nglob >> 7, ss = nglob & 127;
        rgb_out[col * (SAMPLES * RAYS) + ss * RAYS + rr] = sg;
      }
    }
  }
}

__global__ __launch_bounds__(256) void nerf_integrate_kernel(
    const float* __restrict__ rays_o, const float* __restrict__ rays_d,
    const float* __restrict__ aabb, const float* __restrict__ bg,
    const float* __restrict__ density, const float* __restrict__ rgb,
    float* __restrict__ out)
{
  int r = blockIdx.x * 256 + threadIdx.x;
  if (r >= RAYS) return;
  float dx = rays_d[r * 3 + 0], dy = rays_d[r * 3 + 1], dz = rays_d[r * 3 + 2];
  float rn = rsqrtf(dx * dx + dy * dy + dz * dz);
  dx *= rn; dy *= rn; dz *= rn;
  float ox = rays_o[r * 3 + 0], oy = rays_o[r * 3 + 1], oz = rays_o[r * 3 + 2];
  float a0x = aabb[0], a0y = aabb[1], a0z = aabb[2];
  float a1x = aabb[3], a1y = aabb[4], a1z = aabb[5];
  float ivx = 1.f / dx, ivy = 1.f / dy, ivz = 1.f / dz;
  float tx0 = (a0x - ox) * ivx, tx1 = (a1x - ox) * ivx;
  float ty0 = (a0y - oy) * ivy, ty1 = (a1y - oy) * ivy;
  float tz0 = (a0z - oz) * ivz, tz1 = (a1z - oz) * ivz;
  float nearv = fmaxf(fmaxf(fminf(tx0, tx1), fminf(ty0, ty1)), fminf(tz0, tz1));
  nearv = fmaxf(nearv, 0.f);
  float farv = fminf(fminf(fmaxf(tx0, tx1), fmaxf(ty0, ty1)), fmaxf(tz0, tz1));
  farv = fmaxf(farv, nearv);
  float delta = (farv - nearv) * (1.f / (float)SAMPLES);

  float T = 1.f, asum = 0.f, c0 = 0.f, c1 = 0.f, c2 = 0.f;
  for (int s = 0; s < SAMPLES; ++s) {
    float tau = density[s * RAYS + r] * delta;
    float e = __expf(-tau);
    float w = T * (1.f - e);
    c0 += w * rgb[0 * SAMPLES * RAYS + s * RAYS + r];
    c1 += w * rgb[1 * SAMPLES * RAYS + s * RAYS + r];
    c2 += w * rgb[2 * SAMPLES * RAYS + s * RAYS + r];
    asum += w;
    T *= e;
  }
  float bgc = bg[0];
  out[r * 3 + 0] = c0 + (1.f - asum) * bgc;
  out[r * 3 + 1] = c1 + (1.f - asum) * bgc;
  out[r * 3 + 2] = c2 + (1.f - asum) * bgc;
}

extern "C" void kernel_launch(void* const* d_in, const int* in_sizes, int n_in,
                              void* d_out, int out_size, void* d_ws, size_t ws_size,
                              hipStream_t stream) {
  (void)in_sizes; (void)n_in; (void)out_size; (void)ws_size;
  const float* rays_o = (const float*)d_in[0];
  const float* rays_d = (const float*)d_in[1];
  const float* bg     = (const float*)d_in[2];
  const float* p01    = (const float*)d_in[3];
  const float* p02    = (const float*)d_in[4];
  const float* p12    = (const float*)d_in[5];
  const float* fgrid  = (const float*)d_in[6];
  const float* w1     = (const float*)d_in[7];
  const float* b1     = (const float*)d_in[8];
  const float* w2     = (const float*)d_in[9];
  const float* b2     = (const float*)d_in[10];
  const float* wc1    = (const float*)d_in[11];
  const float* bc1    = (const float*)d_in[12];
  const float* wc2    = (const float*)d_in[13];
  const float* bc2    = (const float*)d_in[14];
  const float* aabb   = (const float*)d_in[15];

  float* density_ws = (float*)d_ws;                    // [S][R]
  float* rgb_ws     = density_ws + RAYS * SAMPLES;     // [3][S][R]
  float* out        = (float*)d_out;

  const int N = RAYS * SAMPLES;
  nerf_points_kernel<<<N / 256, 256, 0, stream>>>(
      rays_o, rays_d, p01, p02, p12, fgrid,
      w1, b1, w2, b2, wc1, bc1, wc2, bc2, aabb,
      density_ws, rgb_ws);
  nerf_integrate_kernel<<<(RAYS + 255) / 256, 256, 0, stream>>>(
      rays_o, rays_d, aabb, bg, density_ws, rgb_ws, out);
}
